// UniversalLQRTransformer_22943715295668
// MI455X (gfx1250) — compile-verified, matrix operand order checked by reading, not run on hardware
//
#include <hip/hip_runtime.h>
#include <hip/hip_bf16.h>
#include <math.h>

// Problem constants (match reference)
#define B_   4
#define S_   1024
#define D_   1024
#define H_   16
#define L_   6
#define DFF_ 4096
#define IN_  17
#define MC_  8
#define DK_  64

typedef __attribute__((ext_vector_type(16))) __bf16 v16bf;
typedef __attribute__((ext_vector_type(8)))  __bf16 v8bf;
typedef __attribute__((ext_vector_type(8)))  float  v8f;
typedef unsigned int uint32x4 __attribute__((ext_vector_type(4)));
typedef int          int32x4  __attribute__((ext_vector_type(4)));
typedef int          int32x8  __attribute__((ext_vector_type(8)));

// fp32 -> bf16 round-to-nearest-even (bit ops only; no cvt dependency)
static __device__ inline __bf16 f2bf(float f) {
  unsigned u = __builtin_bit_cast(unsigned, f);
  u += 0x7FFFu + ((u >> 16) & 1u);
  unsigned short s = (unsigned short)(u >> 16);
  return __builtin_bit_cast(__bf16, s);
}
static __device__ inline float bf2f(__bf16 h) {
  unsigned u = ((unsigned)__builtin_bit_cast(unsigned short, h)) << 16;
  return __builtin_bit_cast(float, u);
}

union VB16 { v16bf v; v8bf h[2]; };

// Assemble a 16x32 bf16 WMMA A-fragment (also valid for B^T stored
// row-major-by-n) from an LDS tile. ISA 16-bit layout: lanes 0-15 hold
// K {0..7,16..23}, lanes 16-31 hold K {8..15,24..31}. 2x ds_load_b128.
static __device__ inline v16bf load_frag(const __bf16* base, int row, int stride,
                                         int kbase, int lane) {
  int koff = (lane & 16) ? 8 : 0;
  const __bf16* p = base + row * stride + kbase + koff;
  VB16 u;
  u.h[0] = *(const v8bf*)(p);
  u.h[1] = *(const v8bf*)(p + 16);
  return u.v;
}

static __device__ inline float gelu_f(float x) {
  return 0.5f * x * (1.0f + erff(x * 0.70710678118f));
}

// ---------------------------------------------------------------------------
// Tensor Data Mover: async 2D bf16 tile (tile_k x tile_rows) from global to
// LDS, row stride row_stride_elems. D# built per CDNA5 ISA ch.8 (group0:
// count/lds/global/type=2; group1: data_size=2B, dims, stride). Tensor dims
// declared == tile dims (tiles never straddle bounds here), real row stride.
// ---------------------------------------------------------------------------
static __device__ inline void tdm_load_2d(unsigned lds_off, const void* gaddr,
                                          unsigned tile_k, unsigned tile_rows,
                                          unsigned row_stride_elems) {
  unsigned long long ga = (unsigned long long)(size_t)gaddr;
  uint32x4 g0;
  g0[0] = 1u;                                    // count=1, user mode, no gather
  g0[1] = lds_off;                               // lds_addr (bytes)
  g0[2] = (unsigned)(ga & 0xFFFFFFFFu);          // global_addr[31:0]
  g0[3] = (unsigned)((ga >> 32) & 0x01FFFFFFu) | (2u << 30);  // [56:32] | type=2
  int32x8 g1;
  g1[0] = (int)(1u << 16);                       // data_size=1 -> 2 bytes
  g1[1] = (int)((tile_k & 0xFFFFu) << 16);       // tensor_dim0 lo (abar=0)
  g1[2] = (int)((tile_k >> 16) | ((tile_rows & 0xFFFFu) << 16)); // dim0 hi|dim1 lo
  g1[3] = (int)((tile_rows >> 16) | (tile_k << 16));             // dim1 hi|tile_dim0
  g1[4] = (int)(tile_rows & 0xFFFFu);            // tile_dim1 | tile_dim2=0
  g1[5] = (int)row_stride_elems;                 // tensor_dim0_stride lo
  g1[6] = 0;                                     // stride0 hi | dim1_stride lo
  g1[7] = 0;
  int32x4 z = {0, 0, 0, 0};
#if __clang_major__ >= 23
  int32x8 z8 = {0, 0, 0, 0, 0, 0, 0, 0};
  __builtin_amdgcn_tensor_load_to_lds(g0, g1, z, z, z8, 0);
#else
  __builtin_amdgcn_tensor_load_to_lds(g0, g1, z, z, 0);
#endif
}

// ---------------------------------------------------------------------------
// Weight prep: W[K][N] fp32 -> Wt[N][K] bf16 (layer-stacked via blockIdx.y)
// ---------------------------------------------------------------------------
__global__ __launch_bounds__(256) void convT_kernel(
    const float* __restrict__ W, __bf16* __restrict__ Wt, int K, int N)
{
  const size_t lay = (size_t)blockIdx.y * K * N;
  size_t i = (size_t)blockIdx.x * 256 + threadIdx.x;
  if (i >= (size_t)K * N) return;
  int n = (int)(i % N);
  size_t k = i / N;
  Wt[lay + (size_t)n * K + k] = f2bf(W[lay + i]);
}

// ---------------------------------------------------------------------------
// Tiled WMMA GEMM, TDM-fed, double-buffered LDS.
//   out = epilogue( A_bf16[M,K] * Wt_bf16[N,K]^T + bias (+resid) )
// 256 threads = 8 waves (4x2), block tile 128x64, K-step 32.
// ---------------------------------------------------------------------------
#define GT_M 128
#define GT_N 64
#define GT_K 32

__global__ __launch_bounds__(256) void gemm_bf16_kernel(
    const __bf16* __restrict__ A, const __bf16* __restrict__ Wt,
    const float* __restrict__ bias, const float* __restrict__ resid,
    float* __restrict__ outF, __bf16* __restrict__ outH,
    int M, int N, int K, int act)
{
  __shared__ __align__(16) __bf16 As[2][GT_M][GT_K];   // 2 x 8 KB
  __shared__ __align__(16) __bf16 Bs[2][GT_N][GT_K];   // 2 x 4 KB
  const int tid  = threadIdx.x;
  const int wave = tid >> 5;
  const int lane = tid & 31;
  const int wy = wave >> 1;          // 0..3
  const int wx = wave & 1;           // 0..1
  const int m0 = blockIdx.y * GT_M;
  const int n0 = blockIdx.x * GT_N;
  const int ntiles = K / GT_K;

  if (wave == 0) {   // one TDM issue per block (uniform operands -> SGPR D#)
    tdm_load_2d((unsigned)(size_t)&As[0][0][0], A  + (size_t)m0 * K, GT_K, GT_M, (unsigned)K);
    tdm_load_2d((unsigned)(size_t)&Bs[0][0][0], Wt + (size_t)n0 * K, GT_K, GT_N, (unsigned)K);
  }

  v8f acc[2][2] = {};
  for (int t = 0; t < ntiles; ++t) {
    const int cur = t & 1;
    if (wave == 0) {
      if (t + 1 < ntiles) {  // prefetch next tile via TDM, then wait for current
        tdm_load_2d((unsigned)(size_t)&As[cur ^ 1][0][0],
                    A  + (size_t)m0 * K + (size_t)(t + 1) * GT_K, GT_K, GT_M, (unsigned)K);
        tdm_load_2d((unsigned)(size_t)&Bs[cur ^ 1][0][0],
                    Wt + (size_t)n0 * K + (size_t)(t + 1) * GT_K, GT_K, GT_N, (unsigned)K);
        __builtin_amdgcn_s_wait_tensorcnt(2);   // in-order: first 2 (current tile) done
      } else {
        __builtin_amdgcn_s_wait_tensorcnt(0);
      }
    }
    __syncthreads();
    const __bf16* Ab = &As[cur][0][0];
    const __bf16* Bb = &Bs[cur][0][0];
#pragma unroll
    for (int mi = 0; mi < 2; ++mi) {
      v16bf af = load_frag(Ab, wy * 32 + mi * 16 + (lane & 15), GT_K, 0, lane);
#pragma unroll
      for (int ni = 0; ni < 2; ++ni) {
        v16bf bf_ = load_frag(Bb, wx * 32 + ni * 16 + (lane & 15), GT_K, 0, lane);
        acc[mi][ni] = __builtin_amdgcn_wmma_f32_16x16x32_bf16(
            false, af, false, bf_, (short)0, acc[mi][ni], false, false);
      }
    }
    __syncthreads();
  }

  // Epilogue. C layout: lane holds col n=lane&15; vgpr v holds row v+8*(lane>=16)
  const int half = lane >> 4;
  const int nloc = lane & 15;
#pragma unroll
  for (int mi = 0; mi < 2; ++mi)
#pragma unroll
    for (int ni = 0; ni < 2; ++ni) {
      int gn = n0 + wx * 32 + ni * 16 + nloc;
      float bv = bias[gn];
#pragma unroll
      for (int v = 0; v < 8; ++v) {
        int gm = m0 + wy * 32 + mi * 16 + v + half * 8;
        float val = acc[mi][ni][v] + bv;
        if (act == 1) val = gelu_f(val);
        if (resid) val += resid[(size_t)gm * N + gn];
        size_t idx = (size_t)gm * N + gn;
        if (outF) outF[idx] = val;
        if (outH) outH[idx] = f2bf(val);
      }
    }
}

// ---------------------------------------------------------------------------
// Flash attention, one wave per (b, h, 16-row q-tile). Q/K tiles via TDM
// (bf16, strided rows), V manually transposed into LDS (16B vector loads),
// WMMA for Q*K^T and P*V, online softmax in C-fragment layout.
// ---------------------------------------------------------------------------
__global__ __launch_bounds__(32) void flash_attn_kernel(
    const __bf16* __restrict__ Q, const __bf16* __restrict__ Kp,
    const __bf16* __restrict__ Vp, const int* __restrict__ pad,
    __bf16* __restrict__ O)
{
  __shared__ __align__(16) __bf16 Qt[16][DK_];   // 2 KB
  __shared__ __align__(16) __bf16 Kt[32][DK_];   // 4 KB
  __shared__ __align__(16) __bf16 Vt[DK_][32];   // 4 KB (V^T = B-operand for PV)
  __shared__ __align__(16) __bf16 Pt[16][32];    // 1 KB

  const int lane = threadIdx.x & 31;
  const int qt = blockIdx.x, h = blockIdx.y, b = blockIdx.z;
  const int q0 = qt * 16;
  const size_t baseoff = ((size_t)b * S_) * D_ + (size_t)h * DK_;
  const int half = lane >> 4;
  const int nloc = lane & 15;
  const float scale = 0.125f;  // 1/sqrt(DK)

  // Q tile 16x64 via TDM (single wave: issue once, EXEC-independent)
  tdm_load_2d((unsigned)(size_t)&Qt[0][0], Q + baseoff + (size_t)q0 * D_,
              DK_, 16, D_);

  float mstate[8], lstate[8];
#pragma unroll
  for (int v = 0; v < 8; ++v) { mstate[v] = -3.0e38f; lstate[v] = 0.0f; }
  v8f acc[4] = {};

  for (int kb = 0; kb < q0 + 16; kb += 32) {
    // K tile 32x64 via TDM
    tdm_load_2d((unsigned)(size_t)&Kt[0][0], Kp + baseoff + (size_t)kb * D_,
                DK_, 32, D_);
    // V: each lane copies its row (kb+lane) with 16B loads, transposed store
    {
      const __bf16* vrow = Vp + baseoff + (size_t)(kb + lane) * D_;
#pragma unroll
      for (int c8 = 0; c8 < 8; ++c8) {
        v8bf vv = *(const v8bf*)(vrow + c8 * 8);
#pragma unroll
        for (int j = 0; j < 8; ++j) Vt[c8 * 8 + j][lane] = vv[j];
      }
    }
    __builtin_amdgcn_s_wait_tensorcnt(0);
    __syncthreads();

    // scores = Q * K^T (two n-tiles, DK=64 -> 2 k-steps)
    v8f sf[2] = {};
#pragma unroll
    for (int nt = 0; nt < 2; ++nt)
#pragma unroll
      for (int ks = 0; ks < 2; ++ks) {
        v16bf qf = load_frag(&Qt[0][0], lane & 15, DK_, ks * 32, lane);
        v16bf kf = load_frag(&Kt[0][0], nt * 16 + (lane & 15), DK_, ks * 32, lane);
        sf[nt] = __builtin_amdgcn_wmma_f32_16x16x32_bf16(
            false, qf, false, kf, (short)0, sf[nt], false, false);
      }

    // scale + causal/padding mask + row max
    float rowmax[8];
#pragma unroll
    for (int v = 0; v < 8; ++v) rowmax[v] = -3.0e38f;
#pragma unroll
    for (int nt = 0; nt < 2; ++nt) {
      int kg = kb + nt * 16 + nloc;
      int pm = pad[b * S_ + kg];
#pragma unroll
      for (int v = 0; v < 8; ++v) {
        int mg = q0 + v + half * 8;
        float s = sf[nt][v] * scale;
        s = (kg > mg || pm == 0) ? -1.0e9f : s;
        sf[nt][v] = s;
        rowmax[v] = fmaxf(rowmax[v], s);
      }
    }
#pragma unroll
    for (int off = 1; off < 16; off <<= 1)
#pragma unroll
      for (int v = 0; v < 8; ++v)
        rowmax[v] = fmaxf(rowmax[v], __shfl_xor(rowmax[v], off, 32));

    float escale[8], rowsum[8];
#pragma unroll
    for (int v = 0; v < 8; ++v) {
      float mnew = fmaxf(mstate[v], rowmax[v]);
      escale[v] = __expf(mstate[v] - mnew);
      mstate[v] = mnew;
      rowsum[v] = 0.0f;
    }
    // P = exp(s-m): LDS round-trip re-layouts C-frag -> A-frag
#pragma unroll
    for (int nt = 0; nt < 2; ++nt)
#pragma unroll
      for (int v = 0; v < 8; ++v) {
        float p = __expf(sf[nt][v] - mstate[v]);
        rowsum[v] += p;
        Pt[v + half * 8][nt * 16 + nloc] = f2bf(p);
      }
#pragma unroll
    for (int off = 1; off < 16; off <<= 1)
#pragma unroll
      for (int v = 0; v < 8; ++v)
        rowsum[v] += __shfl_xor(rowsum[v], off, 32);
#pragma unroll
    for (int v = 0; v < 8; ++v) lstate[v] = lstate[v] * escale[v] + rowsum[v];
#pragma unroll
    for (int nt = 0; nt < 4; ++nt)
#pragma unroll
      for (int v = 0; v < 8; ++v) acc[nt][v] *= escale[v];
    __syncthreads();

    // acc += P(16x32) * V(32x64)
    v16bf pf = load_frag(&Pt[0][0], lane & 15, 32, 0, lane);
#pragma unroll
    for (int nt = 0; nt < 4; ++nt) {
      v16bf vf = load_frag(&Vt[0][0], nt * 16 + (lane & 15), 32, 0, lane);
      acc[nt] = __builtin_amdgcn_wmma_f32_16x16x32_bf16(
          false, pf, false, vf, (short)0, acc[nt], false, false);
    }
  }

  // ctx (bf16) = acc / l
#pragma unroll
  for (int nt = 0; nt < 4; ++nt)
#pragma unroll
    for (int v = 0; v < 8; ++v) {
      int mg = q0 + v + half * 8;
      O[baseoff + (size_t)mg * D_ + nt * 16 + nloc] = f2bf(acc[nt][v] / lstate[v]);
    }
}

// ---------------------------------------------------------------------------
// Embedding: x = in @ emb_W + emb_b + pe (K=17; writes fp32 master + bf16 copy)
// ---------------------------------------------------------------------------
__global__ __launch_bounds__(256) void embed_kernel(
    const float* __restrict__ in, const float* __restrict__ W,
    const float* __restrict__ bvec, const float* __restrict__ pe,
    float* __restrict__ x, __bf16* __restrict__ xh)
{
  size_t i = (size_t)blockIdx.x * blockDim.x + threadIdx.x;
  if (i >= (size_t)B_ * S_ * D_) return;
  int d = (int)(i % D_);
  size_t bs = i / D_;
  int s = (int)(bs % S_);
  const float* row = in + bs * IN_;
  float acc = bvec[d] + pe[(size_t)s * D_ + d];
#pragma unroll
  for (int j = 0; j < IN_; ++j) acc += row[j] * W[(size_t)j * D_ + d];
  x[i] = acc;
  xh[i] = f2bf(acc);
}

// ---------------------------------------------------------------------------
// LayerNorm over D per row; writes fp32 master + bf16 copy
// ---------------------------------------------------------------------------
__global__ __launch_bounds__(256) void ln_kernel(
    const float* __restrict__ X, const float* __restrict__ g,
    const float* __restrict__ bt, float* __restrict__ Y, __bf16* __restrict__ Yh)
{
  __shared__ float sbuf[2][8];
  const int row = blockIdx.x;
  const float* x = X + (size_t)row * D_;
  float sum = 0.0f, sq = 0.0f;
  for (int i = threadIdx.x; i < D_; i += 256) { float v = x[i]; sum += v; sq += v * v; }
#pragma unroll
  for (int off = 1; off < 32; off <<= 1) {
    sum += __shfl_xor(sum, off, 32);
    sq  += __shfl_xor(sq,  off, 32);
  }
  int wave = threadIdx.x >> 5, lane = threadIdx.x & 31;
  if (lane == 0) { sbuf[0][wave] = sum; sbuf[1][wave] = sq; }
  __syncthreads();
  if (threadIdx.x == 0) {
    float a = 0.0f, b2 = 0.0f;
    for (int w = 0; w < 8; ++w) { a += sbuf[0][w]; b2 += sbuf[1][w]; }
    sbuf[0][0] = a; sbuf[1][0] = b2;
  }
  __syncthreads();
  float mean = sbuf[0][0] * (1.0f / D_);
  float var  = sbuf[1][0] * (1.0f / D_) - mean * mean;
  float inv  = rsqrtf(var + 1e-5f);
  for (int i = threadIdx.x; i < D_; i += 256) {
    float o = (x[i] - mean) * inv * g[i] + bt[i];
    Y[(size_t)row * D_ + i]  = o;
    Yh[(size_t)row * D_ + i] = f2bf(o);
  }
}

// ---------------------------------------------------------------------------
// Head2: out[b,s,c] = mask[b,c]*(h_bf16 @ head2_W + head2_b); N=8 < WMMA tile
// ---------------------------------------------------------------------------
__global__ __launch_bounds__(256) void head2_kernel(
    const __bf16* __restrict__ hbuf, const float* __restrict__ W,
    const float* __restrict__ bvec, const int* __restrict__ cmask,
    float* __restrict__ out)
{
  int i = blockIdx.x * blockDim.x + threadIdx.x;
  if (i >= B_ * S_ * MC_) return;
  int c = i % MC_;
  int bs = i / MC_;
  int bb = bs / S_;
  const __bf16* hr = hbuf + (size_t)bs * DFF_;
  float acc = bvec[c];
  for (int k = 0; k < DFF_; ++k) acc += bf2f(hr[k]) * W[(size_t)k * MC_ + c];
  out[i] = acc * (float)cmask[bb * MC_ + c];
}

// ---------------------------------------------------------------------------
extern "C" void kernel_launch(void* const* d_in, const int* in_sizes, int n_in,
                              void* d_out, int out_size, void* d_ws, size_t ws_size,
                              hipStream_t stream) {
  (void)in_sizes; (void)n_in; (void)out_size; (void)ws_size;
  const float* in_seq = (const float*)d_in[0];
  const int*   padm   = (const int*)d_in[1];
  const int*   cmask  = (const int*)d_in[2];
  const float* embW   = (const float*)d_in[3];
  const float* embB   = (const float*)d_in[4];
  const float* pe     = (const float*)d_in[5];
  const float* Wq = (const float*)d_in[6];
  const float* Wk = (const float*)d_in[7];
  const float* Wv = (const float*)d_in[8];
  const float* Wo = (const float*)d_in[9];
  const float* bq = (const float*)d_in[10];
  const float* bk = (const float*)d_in[11];
  const float* bv = (const float*)d_in[12];
  const float* bo = (const float*)d_in[13];
  const float* ln1s = (const float*)d_in[14];
  const float* ln1b = (const float*)d_in[15];
  const float* ln2s = (const float*)d_in[16];
  const float* ln2b = (const float*)d_in[17];
  const float* f1W = (const float*)d_in[18];
  const float* f1b = (const float*)d_in[19];
  const float* f2W = (const float*)d_in[20];
  const float* f2b = (const float*)d_in[21];
  const float* h1W = (const float*)d_in[22];
  const float* h1b = (const float*)d_in[23];
  const float* h2W = (const float*)d_in[24];
  const float* h2b = (const float*)d_in[25];

  const int M = B_ * S_;                         // 4096
  const size_t NSD = (size_t)M * D_;             // 4 Mi elems
  const size_t NFF = (size_t)M * DFF_;           // 16 Mi elems
  const size_t WDD = (size_t)D_ * D_;            // 1 Mi
  const size_t WDF = (size_t)D_ * DFF_;          // 4 Mi

  char* p = (char*)d_ws;
  float*  x    = (float*)p;  p += NSD * sizeof(float);
  float*  t0   = (float*)p;  p += NSD * sizeof(float);
  __bf16* xh   = (__bf16*)p; p += NSD * 2;
  __bf16* qh   = (__bf16*)p; p += NSD * 2;
  __bf16* kh   = (__bf16*)p; p += NSD * 2;
  __bf16* vh   = (__bf16*)p; p += NSD * 2;
  __bf16* ctxh = (__bf16*)p; p += NSD * 2;
  __bf16* ffh  = (__bf16*)p; p += NFF * 2;
  __bf16* WqT  = (__bf16*)p; p += L_ * WDD * 2;
  __bf16* WkT  = (__bf16*)p; p += L_ * WDD * 2;
  __bf16* WvT  = (__bf16*)p; p += L_ * WDD * 2;
  __bf16* WoT  = (__bf16*)p; p += L_ * WDD * 2;
  __bf16* f1T  = (__bf16*)p; p += L_ * WDF * 2;
  __bf16* f2T  = (__bf16*)p; p += L_ * WDF * 2;
  __bf16* h1T  = (__bf16*)p; p += WDF * 2;

  const dim3 gD (D_   / GT_N, M / GT_M);         // (16, 32)
  const dim3 gFF(DFF_ / GT_N, M / GT_M);         // (64, 32)
  const dim3 gAt(S_ / 16, H_, B_);               // (64, 16, 4)
  const int gDDb = (int)((WDD + 255) / 256);
  const int gDFb = (int)((WDF + 255) / 256);

  // One-time weight convert+transpose (fp32 -> bf16, [K][N] -> [N][K])
  convT_kernel<<<dim3(gDDb, L_), 256, 0, stream>>>(Wq, WqT, D_, D_);
  convT_kernel<<<dim3(gDDb, L_), 256, 0, stream>>>(Wk, WkT, D_, D_);
  convT_kernel<<<dim3(gDDb, L_), 256, 0, stream>>>(Wv, WvT, D_, D_);
  convT_kernel<<<dim3(gDDb, L_), 256, 0, stream>>>(Wo, WoT, D_, D_);
  convT_kernel<<<dim3(gDFb, L_), 256, 0, stream>>>(f1W, f1T, D_,  DFF_);
  convT_kernel<<<dim3(gDFb, L_), 256, 0, stream>>>(f2W, f2T, DFF_, D_);
  convT_kernel<<<dim3(gDFb, 1 ), 256, 0, stream>>>(h1W, h1T, D_,  DFF_);

  embed_kernel<<<(int)((NSD + 255) / 256), 256, 0, stream>>>(in_seq, embW, embB, pe, x, xh);

  for (int l = 0; l < L_; ++l) {
    const size_t wdd = (size_t)l * WDD;
    const size_t wdf = (size_t)l * WDF;
    const size_t lD  = (size_t)l * D_;
    gemm_bf16_kernel<<<gD, 256, 0, stream>>>(xh, WqT + wdd, bq + lD, nullptr,
                                             nullptr, qh, M, D_, D_, 0);
    gemm_bf16_kernel<<<gD, 256, 0, stream>>>(xh, WkT + wdd, bk + lD, nullptr,
                                             nullptr, kh, M, D_, D_, 0);
    gemm_bf16_kernel<<<gD, 256, 0, stream>>>(xh, WvT + wdd, bv + lD, nullptr,
                                             nullptr, vh, M, D_, D_, 0);
    flash_attn_kernel<<<gAt, 32, 0, stream>>>(qh, kh, vh, padm, ctxh);
    gemm_bf16_kernel<<<gD, 256, 0, stream>>>(ctxh, WoT + wdd, bo + lD, x,
                                             t0, nullptr, M, D_, D_, 0);
    ln_kernel<<<M, 256, 0, stream>>>(t0, ln1s + lD, ln1b + lD, x, xh);
    gemm_bf16_kernel<<<gFF, 256, 0, stream>>>(xh, f1T + wdf, f1b + (size_t)l * DFF_,
                                              nullptr, nullptr, ffh, M, DFF_, D_, 1);
    gemm_bf16_kernel<<<gD, 256, 0, stream>>>(ffh, f2T + wdf, f2b + lD, x,
                                             t0, nullptr, M, D_, DFF_, 0);
    ln_kernel<<<M, 256, 0, stream>>>(t0, ln2s + lD, ln2b + lD, x, xh);
  }

  gemm_bf16_kernel<<<gFF, 256, 0, stream>>>(xh, h1T, h1b, nullptr,
                                            nullptr, ffh, M, DFF_, D_, 1);
  head2_kernel<<<(M * MC_ + 255) / 256, 256, 0, stream>>>(ffh, h2W, h2b, cmask,
                                                          (float*)d_out);
}
